// Attention_28630251995729
// MI455X (gfx1250) — matrix operand order repaired and checked by reading, not compiled, not run on hardware
//
#include <hip/hip_runtime.h>
#include <hip/hip_bf16.h>
#include <math.h>

typedef _Float16 f16;
typedef __attribute__((ext_vector_type(16))) _Float16 v16h;
typedef __attribute__((ext_vector_type(8)))  _Float16 v8h;
typedef __attribute__((ext_vector_type(8)))  float    v8f;
typedef __attribute__((ext_vector_type(4)))  unsigned u32x4;
typedef __attribute__((ext_vector_type(8)))  unsigned u32x8;

#define DIMC   512
#define INNERC 512
#define NSEQ   1024
#define BATCH  4
#define HEADSC 8
#define DH     64
#define NROWS  (HEADSC*BATCH*NSEQ)      // 32768 head-rows
#define GAMMAC 0.01f
#define INVSQD 0.12499998f               // 1/(sqrt(64)+1e-6)

// ---------------- WMMA helpers (CDNA5 v_wmma_f32_16x16x32_f16) ----------------

static __device__ __forceinline__ v8f wmma16(v16h a, v16h b, v8f c) {
  return __builtin_amdgcn_wmma_f32_16x16x32_f16(false, a, false, b, (short)0, c, false, false);
}

// A-matrix 16x32 f16 per-lane layout (contiguous K, matches B loaders): lane<16:
// row=lane, K in kb..kb+15 with kb=0; lanes 16..31: same rows, kb=16.
static __device__ __forceinline__ v16h load_A16(const f16* base, int ld) {
  int lane = threadIdx.x & 31;
  int r  = lane & 15;
  int kb = (lane >> 4) ? 16 : 0;
  const f16* p = base + (size_t)r * ld + kb;
  v8h lo = *(const v8h*)(p);
  v8h hi = *(const v8h*)(p + 8);
  v16h a;
#pragma unroll
  for (int t = 0; t < 8; ++t) { a[t] = lo[t]; a[t + 8] = hi[t]; }
  return a;
}

// B-matrix 32x16 f16, B[k][n] = X[n][k] (B = tile-of-rows^T): one contiguous
// 32B load per lane. lane: n=lane&15, K-halves (lane>>4)*16 .. +15.
static __device__ __forceinline__ v16h load_BT16(const f16* base, int ld) {
  int lane = threadIdx.x & 31;
  int n  = lane & 15;
  int kb = (lane >> 4) * 16;
  return *(const v16h*)(base + (size_t)n * ld + kb);
}

// B-matrix 32x16 from pre-swizzled packed f16 weights: one 32B load per lane.
// Packed layout: tile (k0/32, n0/16) stored as 512 halves, lane-major.
static __device__ __forceinline__ v16h load_B_packed(const f16* Wp, int k0, int n0) {
  int lane = threadIdx.x & 31;
  return *(const v16h*)(Wp + (((size_t)(k0 >> 5) * 32 + (n0 >> 4)) << 9) + lane * 16);
}

// B-matrix 32x16 from row-major f16 V tile: B[k][n] = V[k][n0+n]
static __device__ __forceinline__ v16h load_B_f16(const f16* V, int ld, int n0) {
  int lane = threadIdx.x & 31;
  int n  = n0 + (lane & 15);
  int kb = (lane >> 4) * 16;
  v16h b;
#pragma unroll
  for (int t = 0; t < 16; ++t) b[t] = V[(size_t)(kb + t) * ld + n];
  return b;
}

// ---------------- Tensor Data Mover: 2-D tile (64 x 32 rows, f16) -> LDS -----
// D# per ISA 8.3/8.4: group0 = {count|flags, lds_addr, global_addr, type=2};
// group1 = {data_size=2B, tensor_dim0=64, tensor_dim1=1024, tile=64x32,
//           tensor_dim0_stride=64}. Issued from one lane; EXEC ignored by TDM.

static __device__ __forceinline__ void tdm_load_tile(unsigned lds_off, const f16* gptr) {
  unsigned long long ga = (unsigned long long)(uintptr_t)gptr;
  u32x4 g0;
  g0[0] = 1u;                                        // count=1, user mode
  g0[1] = lds_off;                                   // LDS byte address
  g0[2] = (unsigned)ga;                              // global_addr[31:0]
  g0[3] = (unsigned)((ga >> 32) & 0x1FFFFFFu) | (2u << 30);  // addr[56:32]|type=2
  u32x8 g1;
  g1[0] = 1u << 16;                                  // data_size = 2 bytes
  g1[1] = 64u << 16;                                 // tensor_dim0 = 64
  g1[2] = (unsigned)NSEQ << 16;                      // tensor_dim1 = 1024
  g1[3] = 64u << 16;                                 // tile_dim0 = 64
  g1[4] = 32u;                                       // tile_dim1 = 32 rows
  g1[5] = 64u;                                       // tensor_dim0_stride = 64
  g1[6] = 0u;
  g1[7] = 0u;
  asm volatile("tensor_load_to_lds %0, %1" :: "s"(g0), "s"(g1) : "memory");
}

// ---------------- Kernel 0: pack W_in / W_out into WMMA B-operand layout -----

__global__ __launch_bounds__(256) void pack_w_kernel(
    const float* __restrict__ W_in, const float* __restrict__ W_out,
    f16* __restrict__ Wip, f16* __restrict__ Wop) {
  int idx = blockIdx.x * 256 + threadIdx.x;  // 0 .. 2*512*512-1
  int which = idx >> 18;
  int e     = idx & 0x3FFFF;
  int chunk = e >> 9;                        // (k0/32)*32 + (n0/16)
  int lane  = (e >> 4) & 31;
  int t     = e & 15;
  int k = (chunk >> 5) * 32 + (lane >> 4) * 16 + t;
  int n = (chunk & 31) * 16 + (lane & 15);
  const float* W = which ? W_out : W_in;
  f16* Wp        = which ? Wop : Wip;
  Wp[e] = (f16)W[(size_t)k * 512 + n];
}

// ---------------- Kernel 1: LayerNorm + W_in projection (WMMA GEMM) ----------

__global__ __launch_bounds__(256) void proj_ln_gemm_kernel(
    const float* __restrict__ q, const float* __restrict__ k,
    const float* __restrict__ v, const float* __restrict__ ln_g,
    const float* __restrict__ ln_b, const f16* __restrict__ Wip,
    f16* __restrict__ fq, f16* __restrict__ fk, f16* __restrict__ fv) {
  // grid = 3 * BATCH * (NSEQ/16)
  int bid   = blockIdx.x;
  int ntile = bid & 63;
  int b     = (bid >> 6) & 3;
  int which = bid >> 8;
  const float* src = (which == 0) ? q : (which == 1) ? k : v;
  f16* dst         = (which == 0) ? fq : (which == 1) ? fk : fv;

  __shared__ __attribute__((aligned(32))) f16 sA[16][DIMC];   // 16 KB LN'd rows

  int tid = threadIdx.x;
  int row = tid >> 4;      // 0..15 (each row handled by 16 consecutive lanes)
  int cg  = tid & 15;
  const float* xr = src + ((size_t)b * NSEQ + (size_t)ntile * 16 + row) * DIMC;

  float vals[32];
  float s = 0.f, s2 = 0.f;
#pragma unroll
  for (int j = 0; j < 32; ++j) {
    float x = xr[cg * 32 + j];
    vals[j] = x; s += x; s2 += x * x;
  }
  for (int m = 1; m < 16; m <<= 1) { s += __shfl_xor(s, m); s2 += __shfl_xor(s2, m); }
  float mu  = s / (float)DIMC;
  float var = s2 / (float)DIMC - mu * mu;
  float inv = rsqrtf(var + 1e-5f);
#pragma unroll
  for (int j = 0; j < 32; ++j) {
    int c = cg * 32 + j;
    sA[row][c] = (f16)(((vals[j] - mu) * inv) * ln_g[c] + ln_b[c]);
  }
  __syncthreads();

  int wave = tid >> 5, lane = tid & 31;
  for (int nc = wave * 4; nc < wave * 4 + 4; ++nc) {   // 32 N-chunks / 8 waves
    int n0 = nc * 16;
    v8f acc = {};
    for (int k0 = 0; k0 < DIMC; k0 += 32) {
      v16h a  = load_A16(&sA[0][k0], DIMC);
      v16h bm = load_B_packed(Wip, k0, n0);
      acc = wmma16(a, bm, acc);
    }
    int col = n0 + (lane & 15);
    int h = col >> 6, d = col & 63;
    int rbase = (lane >> 4) * 8;
#pragma unroll
    for (int i = 0; i < 8; ++i) {
      int n = ntile * 16 + rbase + i;
      dst[((((size_t)h * BATCH + b) * NSEQ) + n) * DH + d] = (f16)acc[i];
    }
  }
}

// ---------------- Kernel 2: centering + norms + head means -------------------

__global__ __launch_bounds__(256) void center_q_kernel(
    const f16* __restrict__ fq, f16* __restrict__ fqc,
    float* __restrict__ rownorm_q) {
  int r = blockIdx.x * 256 + threadIdx.x;  // 0..NROWS-1
  const f16* p = fq + (size_t)r * DH;
  float s = 0.f, n2 = 0.f, vv[64];
#pragma unroll 8
  for (int j = 0; j < 64; ++j) { float x = (float)p[j]; vv[j] = x; s += x; n2 += x * x; }
  float mu = s * (1.f / 64.f);
  rownorm_q[r] = sqrtf(n2);
  f16* o = fqc + (size_t)r * DH;
#pragma unroll 8
  for (int j = 0; j < 64; ++j) o[j] = (f16)(vv[j] - mu);
}

__global__ __launch_bounds__(256) void colmean_k_kernel(
    const f16* __restrict__ fk, float* __restrict__ colmean) {
  int idx = blockIdx.x * 256 + threadIdx.x;  // 0..2047 : (h*B)*64 + d
  int hb = idx >> 6, d = idx & 63;
  const f16* p = fk + (size_t)hb * NSEQ * DH + d;
  float s = 0.f;
  for (int n = 0; n < NSEQ; ++n) s += (float)p[(size_t)n * DH];
  colmean[idx] = s * (1.f / (float)NSEQ);
}

__global__ __launch_bounds__(256) void center_k_kernel(
    const f16* __restrict__ fk, const float* __restrict__ colmean,
    f16* __restrict__ fkc, float* __restrict__ rownorm_k) {
  int r = blockIdx.x * 256 + threadIdx.x;  // row 0..NROWS-1
  int hb = r >> 10;
  const f16* p = fk + (size_t)r * DH;
  const float* cm = colmean + (size_t)hb * 64;
  f16* o = fkc + (size_t)r * DH;
  float n2 = 0.f;
#pragma unroll 8
  for (int j = 0; j < 64; ++j) {
    float x = (float)p[j];
    n2 += x * x;
    o[j] = (f16)(x - cm[j]);
  }
  rownorm_k[r] = sqrtf(n2);
}

__global__ __launch_bounds__(256) void headmean_kernel(
    const f16* __restrict__ fq, const f16* __restrict__ fk,
    float* __restrict__ hmq, float* __restrict__ hmk) {
  int idx = blockIdx.x * 256 + threadIdx.x;   // 0..1023
  int which = idx >> 9, h = (idx >> 6) & 7, d = idx & 63;
  const f16* p = (which ? fk : fq) + (size_t)h * BATCH * NSEQ * DH + d;
  float s = 0.f;
  for (int i = 0; i < BATCH * NSEQ; ++i) s += (float)p[(size_t)i * DH];
  (which ? hmk : hmq)[h * 64 + d] = s * (1.f / (float)(BATCH * NSEQ));
}

// ---------------- Kernel 3: tiny per-head weight-predictor MLP ---------------

__global__ __launch_bounds__(32) void mlp_kernel(
    const float* __restrict__ hmq, const float* __restrict__ hmk,
    const float* __restrict__ W1, const float* __restrict__ b1,
    const float* __restrict__ lng, const float* __restrict__ lnb,
    const float* __restrict__ W2, const float* __restrict__ b2,
    const float* __restrict__ W3, const float* __restrict__ b3,
    const float* __restrict__ w_temp, float* __restrict__ wcoef) {
  int h = threadIdx.x;
  if (h >= HEADSC) return;
  float feat[128], h1[192], h2[128];
  for (int d = 0; d < 64; ++d) { feat[d] = hmq[h * 64 + d]; feat[64 + d] = hmk[h * 64 + d]; }
  float s = 0.f, s2 = 0.f;
  for (int j = 0; j < 192; ++j) {
    float a = b1[j];
    for (int i = 0; i < 128; ++i) a += feat[i] * W1[i * 192 + j];
    h1[j] = a; s += a; s2 += a * a;
  }
  float mu = s / 192.f, var = s2 / 192.f - mu * mu, inv = rsqrtf(var + 1e-5f);
  for (int j = 0; j < 192; ++j) {
    float a = (h1[j] - mu) * inv * lng[j] + lnb[j];
    h1[j] = fmaxf(a, 0.f);
  }
  for (int l = 0; l < 128; ++l) {
    float a = b2[l];
    for (int j = 0; j < 192; ++j) a += h1[j] * W2[j * 128 + l];
    h2[l] = fmaxf(a, 0.f);
  }
  float lg[3];
  for (int c = 0; c < 3; ++c) {
    float a = b3[c];
    for (int l = 0; l < 128; ++l) a += h2[l] * W3[l * 3 + c];
    lg[c] = a;
  }
  float mx = fmaxf(lg[0], fmaxf(lg[1], lg[2])), den = 0.f, p[3];
  for (int c = 0; c < 3; ++c) { p[c] = __expf(lg[c] - mx); den += p[c]; }
  float wt = fminf(fmaxf(w_temp[0], 0.05f), 3.0f);
  float q2[3]; float mx2 = -1e30f;
  for (int c = 0; c < 3; ++c) { q2[c] = (p[c] / den) / wt; mx2 = fmaxf(mx2, q2[c]); }
  float den2 = 0.f;
  for (int c = 0; c < 3; ++c) { q2[c] = __expf(q2[c] - mx2); den2 += q2[c]; }
  float wsum = 0.f, w[3];
  for (int c = 0; c < 3; ++c) { w[c] = fminf(fmaxf(q2[c] / den2, 0.05f), 0.85f); wsum += w[c]; }
  for (int c = 0; c < 3; ++c) wcoef[h * 3 + c] = w[c] / wsum;
}

// ---------------- Kernel 4: pass-1 score statistics (WMMA) -------------------

__global__ __launch_bounds__(256) void pass1_stats_kernel(
    const f16* __restrict__ fq, const f16* __restrict__ fk,
    const f16* __restrict__ fqc, const f16* __restrict__ fkc,
    const float* __restrict__ rownorm_q, const float* __restrict__ rownorm_k,
    float* __restrict__ rs_cos, float* __restrict__ rs_cov,
    float* __restrict__ rs_margin, float* __restrict__ headsums) {
  int wid   = blockIdx.x * 8 + (threadIdx.x >> 5);     // 2048 waves
  int ntile = wid & 63, b = (wid >> 6) & 3, h = wid >> 8;
  int lane  = threadIdx.x & 31;
  size_t hb = (size_t)h * BATCH + b;

  const f16* fqb  = fq  + (hb * NSEQ + (size_t)ntile * 16) * DH;
  const f16* fqcb = fqc + (hb * NSEQ + (size_t)ntile * 16) * DH;
  v16h aq0 = load_A16(fqb, DH),      aq1 = load_A16(fqb + 32, DH);
  v16h ac0 = load_A16(fqcb, DH),     ac1 = load_A16(fqcb + 32, DH);

  int rbase = (lane >> 4) * 8;
  float nq[8];
#pragma unroll
  for (int i = 0; i < 8; ++i)
    nq[i] = rownorm_q[hb * NSEQ + ntile * 16 + rbase + i] + 1e-8f;

  float pc = 0.f, pc2 = 0.f, pv = 0.f, pv2 = 0.f, pcc = 0.f;
  float rm[8], rc[8], rv[8];
#pragma unroll
  for (int i = 0; i < 8; ++i) { rm[i] = 0.f; rc[i] = 0.f; rv[i] = 0.f; }

  for (int m0 = 0; m0 < NSEQ; m0 += 16) {
    const f16* fkb  = fk  + (hb * NSEQ + m0) * DH;
    const f16* fkcb = fkc + (hb * NSEQ + m0) * DH;
    __builtin_prefetch(fkb + 16 * DH, 0, 1);
    __builtin_prefetch(fkcb + 16 * DH, 0, 1);
    v16h bk0 = load_BT16(fkb, DH),  bk1 = load_BT16(fkb + 32, DH);
    v16h bc0 = load_BT16(fkcb, DH), bc1 = load_BT16(fkcb + 32, DH);
    v8f s = {};  s = wmma16(aq0, bk0, s);  s = wmma16(aq1, bk1, s);
    v8f sc = {}; sc = wmma16(ac0, bc0, sc); sc = wmma16(ac1, bc1, sc);
    float nk = rownorm_k[hb * NSEQ + m0 + (lane & 15)] + 1e-8f;
#pragma unroll
    for (int i = 0; i < 8; ++i) {
      float cosv = s[i] / (nq[i] * nk);
      cosv = fminf(fmaxf(cosv, -0.98f), 0.98f);
      float covr = sc[i] * INVSQD;
      float marg = fminf(fmaxf(GAMMAC - cosv, 0.f), 8.f);
      pc += cosv; pc2 += cosv * cosv;
      pv += covr; pv2 += covr * covr;
      pcc += cosv * covr;
      rm[i] += marg; rc[i] += cosv; rv[i] += covr;
    }
  }
  // per-row sums: reduce across the 16 column lanes of each half-wave
#pragma unroll
  for (int i = 0; i < 8; ++i) {
    float a = rm[i], c = rc[i], d = rv[i];
    for (int m = 1; m < 16; m <<= 1) {
      a += __shfl_xor(a, m); c += __shfl_xor(c, m); d += __shfl_xor(d, m);
    }
    if ((lane & 15) == 0) {
      size_t gr = hb * NSEQ + ntile * 16 + rbase + i;
      rs_margin[gr] = a; rs_cos[gr] = c; rs_cov[gr] = d;
    }
  }
  // per-head sums: full-wave reduce then one atomic per quantity
  for (int m = 1; m < 32; m <<= 1) {
    pc += __shfl_xor(pc, m); pc2 += __shfl_xor(pc2, m);
    pv += __shfl_xor(pv, m); pv2 += __shfl_xor(pv2, m);
    pcc += __shfl_xor(pcc, m);
  }
  if (lane == 0) {
    atomicAdd(&headsums[h * 5 + 0], pc);
    atomicAdd(&headsums[h * 5 + 1], pc2);
    atomicAdd(&headsums[h * 5 + 2], pv);
    atomicAdd(&headsums[h * 5 + 3], pv2);
    atomicAdd(&headsums[h * 5 + 4], pcc);
  }
}

// ---------------- Kernel 5: finalize global scalars --------------------------

__global__ __launch_bounds__(256) void finalize_kernel(
    const float* __restrict__ headsums, const float* __restrict__ rs_cos,
    const float* __restrict__ rs_cov, const float* __restrict__ rs_margin,
    const float* __restrict__ wcoef, float* __restrict__ coefs) {
  __shared__ float hs[HEADSC][4];  // Svar, Svar2, Scosvar, Scovvar per head
  if (threadIdx.x < HEADSC * 4) ((float*)hs)[threadIdx.x] = 0.f;
  __syncthreads();
  const float M = (float)NSEQ;
  for (int r = threadIdx.x; r < NROWS; r += 256) {
    int h = r >> 12;
    float var = rs_margin[r] / M;
    atomicAdd(&hs[h][0], var);
    atomicAdd(&hs[h][1], var * var);
    atomicAdd(&hs[h][2], var * rs_cos[r]);
    atomicAdd(&hs[h][3], var * rs_cov[r]);
  }
  __syncthreads();
  if (threadIdx.x == 0) {
    const double Ntot = (double)HEADSC * BATCH * NSEQ * NSEQ;
    double Sc = 0, Sc2 = 0, Sv = 0, Sv2 = 0, Svar = 0, Svar2 = 0;
    for (int h = 0; h < HEADSC; ++h) {
      Sc += headsums[h * 5 + 0]; Sc2 += headsums[h * 5 + 1];
      Sv += headsums[h * 5 + 2]; Sv2 += headsums[h * 5 + 3];
      Svar += hs[h][0]; Svar2 += hs[h][1];
    }
    float health = (float)sqrt(fmax(0.0, (Sv2 - Sv * Sv / Ntot) / (Ntot - 1.0)));
    float base = 0.001f / (float)NSEQ;
    float reg  = (health < 1e-5f) ? base * 8.f : (health < 1e-3f) ? base * 3.f : base;
    float cos_norm = (float)sqrt(fmax(0.0, (Sc2 - Sc * Sc / Ntot) / (Ntot - 1.0))) + 1e-6f;
    float cov_norm = reg * health + 1e-6f;   // clip(±30) never binds at this scale
    double SvT = (double)M * Svar, Sv2T = (double)M * Svar2;
    float var_norm = (float)sqrt(fmax(0.0, (Sv2T - SvT * SvT / Ntot) / (Ntot - 1.0))) + 1e-6f;
    float cos_h = fminf(cos_norm, 1.2f);
    float cov_h = fminf(cov_norm * 12.f, 1.2f);
    float var_h = fminf(var_norm * 12.f, 1.2f);

    double SD = 0, SD2 = 0;
    float A[HEADSC], Bc[HEADSC];
    for (int h = 0; h < HEADSC; ++h) {
      float w0 = wcoef[h * 3 + 0], w1 = wcoef[h * 3 + 1], w2 = wcoef[h * 3 + 2];
      A[h]  = w0 * cos_h / cos_norm;
      Bc[h] = w1 * 0.5f * cov_h * reg / cov_norm;   // multiplies raw cov
      float C = w2 * 0.5f * var_h / var_norm;
      double Hc = headsums[h * 5 + 0], Hc2 = headsums[h * 5 + 1];
      double Hv = headsums[h * 5 + 2], Hv2 = headsums[h * 5 + 3];
      double Hcc = headsums[h * 5 + 4];
      SD  += A[h] * Hc + Bc[h] * Hv + (double)C * M * hs[h][0];
      SD2 += (double)A[h] * A[h] * Hc2 + (double)Bc[h] * Bc[h] * Hv2
           + (double)C * C * M * hs[h][1]
           + 2.0 * A[h] * Bc[h] * Hcc
           + 2.0 * A[h] * C * hs[h][2]
           + 2.0 * Bc[h] * C * hs[h][3];
    }
    float div = (float)sqrt(fmax(0.0, (SD2 - SD * SD / Ntot) / (Ntot - 1.0)));
    float temp = (div < 5e-6f) ? 0.03f : (div < 5e-4f) ? 0.15f : 0.4f;
    for (int h = 0; h < HEADSC; ++h) {
      coefs[h]          = A[h] / temp;
      coefs[HEADSC + h] = Bc[h] / temp;
    }
  }
}

// ---------------- Kernel 6: pass-2 flash attention ---------------------------
// All 8 waves of a block share one (h,b) panel; K/Kc/V tiles (32 rows) are
// staged into LDS by the Tensor Data Mover, double-buffered with
// s_wait_tensorcnt so the next chunk's DMA overlaps this chunk's WMMA work.
// Per-row-constant var term dropped: softmax is invariant to uniform row shift.

__global__ __launch_bounds__(256) void pass2_attn_kernel(
    const f16* __restrict__ fq, const f16* __restrict__ fk,
    const f16* __restrict__ fv, const f16* __restrict__ fqc,
    const f16* __restrict__ fkc, const float* __restrict__ rownorm_q,
    const float* __restrict__ rownorm_k, const float* __restrict__ coefs,
    f16* __restrict__ attn16) {
  __shared__ __attribute__((aligned(32))) f16 sK [2][32][DH];   // 8 KB
  __shared__ __attribute__((aligned(32))) f16 sKc[2][32][DH];   // 8 KB
  __shared__ __attribute__((aligned(32))) f16 sV [2][32][DH];   // 8 KB
  __shared__ __attribute__((aligned(32))) f16 sP [8][16][32];   // 8 KB P staging

  int wid   = blockIdx.x * 8 + (threadIdx.x >> 5);
  int ntile = wid & 63, b = (wid >> 6) & 3, h = wid >> 8;   // (h,b) uniform/block
  int lane  = threadIdx.x & 31;
  int wslot = threadIdx.x >> 5;
  size_t hb = (size_t)h * BATCH + b;

  const f16* kpan  = fk  + hb * NSEQ * DH;
  const f16* kcpan = fkc + hb * NSEQ * DH;
  const f16* vpan  = fv  + hb * NSEQ * DH;

  const f16* fqb  = fq  + (hb * NSEQ + (size_t)ntile * 16) * DH;
  const f16* fqcb = fqc + (hb * NSEQ + (size_t)ntile * 16) * DH;
  v16h aq0 = load_A16(fqb, DH),  aq1 = load_A16(fqb + 32, DH);
  v16h ac0 = load_A16(fqcb, DH), ac1 = load_A16(fqcb + 32, DH);

  int rbase = (lane >> 4) * 8;
  float nq[8];
#pragma unroll
  for (int i = 0; i < 8; ++i)
    nq[i] = rownorm_q[hb * NSEQ + ntile * 16 + rbase + i] + 1e-8f;

  float ah = coefs[h], bh = coefs[HEADSC + h];
  float rmax[8], rsum[8];
#pragma unroll
  for (int i = 0; i < 8; ++i) { rmax[i] = -1e30f; rsum[i] = 0.f; }
  v8f O[4];
#pragma unroll
  for (int c = 0; c < 4; ++c) O[c] = {};

  // TDM: prime buffer 0 (3 descriptors: K, Kc, V)
  if (threadIdx.x == 0) {
    tdm_load_tile((unsigned)(uintptr_t)&sK [0][0][0], kpan);
    tdm_load_tile((unsigned)(uintptr_t)&sKc[0][0][0], kcpan);
    tdm_load_tile((unsigned)(uintptr_t)&sV [0][0][0], vpan);
  }

  auto score16 = [&](const f16* kb, const f16* kcb, int mglob, float* l) {
    v16h bk0 = load_BT16(kb, DH),  bk1 = load_BT16(kb + 32, DH);
    v16h bc0 = load_BT16(kcb, DH), bc1 = load_BT16(kcb + 32, DH);
    v8f s = {};  s = wmma16(aq0, bk0, s);  s = wmma16(aq1, bk1, s);
    v8f sc = {}; sc = wmma16(ac0, bc0, sc); sc = wmma16(ac1, bc1, sc);
    float nk = rownorm_k[hb * NSEQ + mglob + (lane & 15)] + 1e-8f;
#pragma unroll
    for (int i = 0; i < 8; ++i) {
      float cosv = s[i] / (nq[i] * nk);
      cosv = fminf(fmaxf(cosv, -0.98f), 0.98f);
      l[i] = ah * cosv + bh * (sc[i] * INVSQD);
    }
  };

  for (int m0 = 0; m0 < NSEQ; m0 += 32) {
    int buf = (m0 >> 5) & 1;
    if (threadIdx.x == 0) {
      if (m0 + 32 < NSEQ) {
        // kick the next chunk's DMA, then wait for the current chunk only
        // (TDM ops from one wave complete in order).
        int nxt = buf ^ 1;
        tdm_load_tile((unsigned)(uintptr_t)&sK [nxt][0][0], kpan  + (m0 + 32) * DH);
        tdm_load_tile((unsigned)(uintptr_t)&sKc[nxt][0][0], kcpan + (m0 + 32) * DH);
        tdm_load_tile((unsigned)(uintptr_t)&sV [nxt][0][0], vpan  + (m0 + 32) * DH);
        __builtin_amdgcn_s_wait_tensorcnt(3);
      } else {
        __builtin_amdgcn_s_wait_tensorcnt(0);
      }
    }
    __syncthreads();

    float l0[8], l1[8];
    score16(&sK[buf][0][0],  &sKc[buf][0][0],  m0,      l0);
    score16(&sK[buf][16][0], &sKc[buf][16][0], m0 + 16, l1);

    float rowscale[8];
#pragma unroll
    for (int i = 0; i < 8; ++i) {
      float t = fmaxf(l0[i], l1[i]);
      for (int m = 1; m < 16; m <<= 1) t = fmaxf(t, __shfl_xor(t, m));
      float nm = fmaxf(rmax[i], t);
      float sc = __expf(rmax[i] - nm);
      rmax[i] = nm;
      rsum[i] *= sc;
      rowscale[i] = sc;
      float p0 = __expf(l0[i] - nm), p1 = __expf(l1[i] - nm);
      float ps = p0 + p1;
      for (int m = 1; m < 16; m <<= 1) ps += __shfl_xor(ps, m);
      rsum[i] += ps;
      sP[wslot][rbase + i][lane & 15]        = (f16)p0;
      sP[wslot][rbase + i][16 + (lane & 15)] = (f16)p1;
    }
#pragma unroll
    for (int c = 0; c < 4; ++c)
#pragma unroll
      for (int i = 0; i < 8; ++i) O[c][i] *= rowscale[i];

    v16h ap = load_A16(&sP[wslot][0][0], 32);
#pragma unroll
    for (int c = 0; c < 4; ++c) {
      v16h bv = load_B_f16(&sV[buf][0][0], DH, c * 16);
      O[c] = wmma16(ap, bv, O[c]);
    }
    __syncthreads();
  }

  // normalize and scatter into [B][N][h*64+d] fp16 for the output GEMM
#pragma unroll
  for (int c = 0; c < 4; ++c) {
    int col = h * 64 + c * 16 + (lane & 15);
#pragma unroll
    for (int i = 0; i < 8; ++i) {
      int n = ntile * 16 + rbase + i;
      attn16[(((size_t)b * NSEQ) + n) * INNERC + col] = (f16)(O[c][i] / rsum[i]);
    }
  }
}

// ---------------- Kernel 7: output projection (WMMA GEMM + bias) -------------

__global__ __launch_bounds__(256) void outproj_kernel(
    const f16* __restrict__ attn16, const f16* __restrict__ Wop,
    const float* __restrict__ b_out, float* __restrict__ out) {
  int ntile = blockIdx.x;                 // 256 tiles of 16 rows over 4096 rows
  int wave = threadIdx.x >> 5, lane = threadIdx.x & 31;
  const f16* Abase = attn16 + (size_t)ntile * 16 * INNERC;
  for (int nc = wave * 4; nc < wave * 4 + 4; ++nc) {
    int n0 = nc * 16;
    v8f acc = {};
    for (int k0 = 0; k0 < INNERC; k0 += 32) {
      v16h a  = load_A16(Abase + k0, INNERC);
      v16h bm = load_B_packed(Wop, k0, n0);
      acc = wmma16(a, bm, acc);
    }
    int col = n0 + (lane & 15);
    float bias = b_out[col];
    int rbase = (lane >> 4) * 8;
#pragma unroll
    for (int i = 0; i < 8; ++i)
      out[((size_t)ntile * 16 + rbase + i) * DIMC + col] = acc[i] + bias;
  }
}

// ---------------- Launch ------------------------------------------------------

extern "C" void kernel_launch(void* const* d_in, const int* in_sizes, int n_in,
                              void* d_out, int out_size, void* d_ws, size_t ws_size,
                              hipStream_t stream) {
  const float* q      = (const float*)d_in[0];
  const float* k      = (const float*)d_in[1];
  const float* v      = (const float*)d_in[2];
  const float* ln_g   = (const float*)d_in[3];
  const float* ln_b   = (const float*)d_in[4];
  const float* W_in   = (const float*)d_in[5];
  const float* wp_W1  = (const float*)d_in[6];
  const float* wp_b1  = (const float*)d_in[7];
  const float* wp_lng = (const float*)d_in[8];
  const float* wp_lnb = (const float*)d_in[9];
  const float* wp_W2  = (const float*)d_in[10];
  const float* wp_b2  = (const float*)d_in[11];
  const float* wp_W3  = (const float*)d_in[12];
  const float* wp_b3  = (const float*)d_in[13];
  const float* w_temp = (const float*)d_in[14];
  const float* W_out  = (const float*)d_in[15];
  const float* b_out  = (const float*)d_in[16];

  const size_t SZT = (size_t)HEADSC * BATCH * NSEQ * DH * sizeof(f16);  // 4 MiB
  char* ws = (char*)d_ws;
  f16* fq     = (f16*)(ws + 0 * SZT);
  f16* fk     = (f16*)(ws + 1 * SZT);
  f16* fv     = (f16*)(ws + 2 * SZT);
  f16* fqc    = (f16*)(ws + 3 * SZT);
  f16* fkc    = (f16*)(ws + 4 * SZT);
  f16* attn16 = (f16*)(ws + 5 * SZT);                  // [B][N][512] fp16
  char* fb = ws + 6 * SZT;
  f16* Wip = (f16*)(fb);                                 fb += (size_t)512 * 512 * sizeof(f16);
  f16* Wop = (f16*)(fb);                                 fb += (size_t)512 * 512 * sizeof(f16);
  float* rownorm_q = (float*)(fb);                       fb += NROWS * sizeof(float);
  float* rownorm_k = (float*)(fb);                       fb += NROWS * sizeof(float);
  float* rs_cos    = (float*)(fb);                       fb += NROWS * sizeof(float);
  float* rs_cov    = (float*)(fb);                       fb += NROWS * sizeof(float);
  float* rs_margin = (float*)(fb);                       fb += NROWS * sizeof(float);
  float* colmean   = (float*)(fb);                       fb += HEADSC * BATCH * 64 * sizeof(float);
  float* hmq       = (float*)(fb);                       fb += HEADSC * 64 * sizeof(float);
  float* hmk       = (float*)(fb);                       fb += HEADSC * 64 * sizeof(float);
  float* headsums  = (float*)(fb);                       fb += HEADSC * 5 * sizeof(float);
  float* wcoef     = (float*)(fb);                       fb += HEADSC * 3 * sizeof(float);
  float* coefs     = (float*)(fb);                       fb += HEADSC * 2 * sizeof(float);

  hipMemsetAsync(headsums, 0, HEADSC * 5 * sizeof(float), stream);

  pack_w_kernel<<<2048, 256, 0, stream>>>(W_in, W_out, Wip, Wop);
  proj_ln_gemm_kernel<<<3 * BATCH * (NSEQ / 16), 256, 0, stream>>>(
      q, k, v, ln_g, ln_b, Wip, fq, fk, fv);
  center_q_kernel<<<NROWS / 256, 256, 0, stream>>>(fq, fqc, rownorm_q);
  colmean_k_kernel<<<(HEADSC * BATCH * 64) / 256, 256, 0, stream>>>(fk, colmean);
  center_k_kernel<<<NROWS / 256, 256, 0, stream>>>(fk, colmean, fkc, rownorm_k);
  headmean_kernel<<<4, 256, 0, stream>>>(fq, fk, hmq, hmk);
  mlp_kernel<<<1, 32, 0, stream>>>(hmq, hmk, wp_W1, wp_b1, wp_lng, wp_lnb,
                                   wp_W2, wp_b2, wp_W3, wp_b3, w_temp, wcoef);
  pass1_stats_kernel<<<256, 256, 0, stream>>>(fq, fk, fqc, fkc, rownorm_q,
                                              rownorm_k, rs_cos, rs_cov,
                                              rs_margin, headsums);
  finalize_kernel<<<1, 256, 0, stream>>>(headsums, rs_cos, rs_cov, rs_margin,
                                         wcoef, coefs);
  pass2_attn_kernel<<<256, 256, 0, stream>>>(fq, fk, fv, fqc, fkc, rownorm_q,
                                             rownorm_k, coefs, attn16);
  outproj_kernel<<<256, 256, 0, stream>>>(attn16, Wop, b_out, (float*)d_out);
}
